// FeatureEncoder_18270790877759
// MI455X (gfx1250) — matrix operand order repaired and checked by reading, not compiled, not run on hardware
//
#include <hip/hip_runtime.h>
#include <hip/hip_bf16.h>

// ---------------------------------------------------------------------------
// wav2vec2-style feature encoder for MI455X (gfx1250), wave32 + WMMA bf16.
// Activations: bf16, time-major [b][t][c] (c contiguous) so WMMA B-operand
// loads and D-operand stores are contiguous 32B/16B per lane.
// Weights: bf16 [co][K], K = dk*512 + ci, staged per-block in LDS via
// GLOBAL_LOAD_ASYNC_TO_LDS_B128 when available (ASYNCcnt), else uint4 copies.
// Each wave computes 16co x 64t (4 accumulators, A reused 4x per K-step).
// ---------------------------------------------------------------------------

typedef __attribute__((ext_vector_type(16))) __bf16 v16bf;
typedef __attribute__((ext_vector_type(8)))  float  v8f;

union BfPack { uint4 q[2]; v16bf v; };

__device__ __forceinline__ unsigned short f2bf(float f) {
    unsigned int u = __float_as_uint(f);
    unsigned int r = u + 0x7FFFu + ((u >> 16) & 1u);   // round-to-nearest-even
    return (unsigned short)(r >> 16);
}
__device__ __forceinline__ float bf2f(unsigned short h) {
    return __uint_as_float(((unsigned int)h) << 16);
}

__device__ __forceinline__ unsigned int hash3(unsigned a, unsigned b, unsigned c) {
    unsigned h = a * 0x9E3779B1u ^ b * 0x85EBCA77u ^ c * 0xC2B2AE3Du;
    h ^= h >> 16; h *= 0x7FEB352Du; h ^= h >> 15; h *= 0x846CA68Bu; h ^= h >> 16;
    return h;
}
__device__ __forceinline__ float u01(unsigned h) { return (float)(h >> 8) * (1.0f / 16777216.0f); }

// ----------------------------- constants -----------------------------------
#define BATCH   8
#define CCH     512
#define F_OUT   501
#define NUM_M   3      // int(501*0.065/10)
#define M_MAX   30
#define K_NEG   100
#define KT_MAX  1536   // 512*3

// d_out flat offsets (return order: masked_features, features, mask, negatives)
#define OFF_MASKED 0
#define OFF_FEAT   2052096
#define OFF_MASK   4104192
#define OFF_NEG    4108200

#if defined(__has_builtin)
#if __has_builtin(__builtin_amdgcn_global_load_async_to_lds_b128)
#define HAVE_ASYNC_LDS 1
#endif
#endif

// ----------------------------- tiny utils ----------------------------------
__global__ void k_zero_f32(float* p, int n) {
    int i = blockIdx.x * blockDim.x + threadIdx.x;
    if (i < n) p[i] = 0.0f;
}

// global mean / sumsq over raw input x (1,280,000 f32)
__global__ void k_reduce_x(const float* __restrict__ x, float* __restrict__ st, long long N) {
    __shared__ float ss[256], sq[256];
    float s = 0.f, q = 0.f;
    for (long long i = (long long)blockIdx.x * blockDim.x + threadIdx.x; i < N;
         i += (long long)gridDim.x * blockDim.x) {
        float v = x[i]; s += v; q += v * v;
    }
    ss[threadIdx.x] = s; sq[threadIdx.x] = q;
    __syncthreads();
    for (int o = 128; o > 0; o >>= 1) {
        if ((int)threadIdx.x < o) { ss[threadIdx.x] += ss[threadIdx.x + o]; sq[threadIdx.x] += sq[threadIdx.x + o]; }
        __syncthreads();
    }
    if (threadIdx.x == 0) { atomicAdd(&st[0], ss[0]); atomicAdd(&st[1], sq[0]); }
}

// ----------------------- layer 1: cin=1, k=10, s=5, p=5 --------------------
__global__ void k_conv1(const float* __restrict__ x, const float* __restrict__ w,
                        const float* __restrict__ bias, const float* __restrict__ st,
                        unsigned short* __restrict__ out) {
    const int T = 32001;
    const long long N = (long long)BATCH * T * CCH;
    long long i = (long long)blockIdx.x * blockDim.x + threadIdx.x;
    if (i >= N) return;
    const float Nx = 1280000.0f;
    float sum = st[0], sq = st[1];
    float mean = sum / Nx;
    float sd = sqrtf(fmaxf((sq - sum * sum / Nx) / (Nx - 1.0f), 0.0f));
    float rinv = 1.0f / (sd + 1e-8f);
    int co = (int)(i & 511);
    long long r = i >> 9;
    int t = (int)(r % T);
    int b = (int)(r / T);
    float acc = bias[co];
#pragma unroll
    for (int dk = 0; dk < 10; ++dk) {
        int pos = t * 5 - 5 + dk;
        if (pos >= 0 && pos < 160000)
            acc += w[co * 10 + dk] * ((x[(size_t)b * 160000 + pos] - mean) * rinv);
    }
    out[i] = f2bf(acc);    // [b][t][co] with co fastest == flat i
}

// -------- weight relayout: (512, 512, k) f32 -> bf16 [co][dk*512+ci] -------
__global__ void k_wprep(const float* __restrict__ w, unsigned short* __restrict__ wb,
                        int K, int KT) {
    int N = CCH * KT;
    int i = blockIdx.x * blockDim.x + threadIdx.x;
    if (i >= N) return;
    int co = i / KT;
    int rem = i % KT;
    int dk = rem >> 9;
    int ci = rem & 511;
    wb[i] = f2bf(w[((size_t)co * CCH + ci) * K + dk]);
}

// -------------------- implicit-GEMM conv via WMMA bf16 ---------------------
// block: 256 threads = 8 waves. blockIdx.y -> 16-co tile, blockIdx.z -> batch,
// each wave -> four 16-column t tiles (A operand reused 4x per K-step).
// Weights tile (16 x KT bf16, <=48KB) staged in LDS.
#define CONV_WAVES 8
__global__ void __launch_bounds__(256)
k_conv_wmma(const unsigned short* __restrict__ in, const unsigned short* __restrict__ wb,
            const float* __restrict__ bias, unsigned short* __restrict__ out,
            int Lin, int T, int S, int P, int KT) {
    __shared__ __align__(16) unsigned short sW[16 * KT_MAX];   // 48 KB max

    const int cobase = blockIdx.y * 16;
    // stage 16 x KT bf16 weights (row-major, identical layout to wb) in 16B granules
    {
        const int nch = (16 * KT) >> 3;                    // # of 8-short chunks
#if defined(HAVE_ASYNC_LDS) && defined(__HIP_DEVICE_COMPILE__)
        typedef int v4i_t __attribute__((vector_size(16)));
        typedef __attribute__((address_space(1))) v4i_t* g_v4i_p;
        typedef __attribute__((address_space(3))) v4i_t* l_v4i_p;
        for (int cidx = threadIdx.x; cidx < nch; cidx += blockDim.x) {
            int idx = cidx << 3;
            __builtin_amdgcn_global_load_async_to_lds_b128(
                (g_v4i_p)(wb + (size_t)cobase * KT + idx),
                (l_v4i_p)&sW[idx],
                /*offset=*/0, /*cpol=*/0);
        }
#if __has_builtin(__builtin_amdgcn_s_wait_asynccnt)
        __builtin_amdgcn_s_wait_asynccnt(0);
#else
        asm volatile("s_wait_asynccnt 0" ::: "memory");
#endif
#else
        for (int cidx = threadIdx.x; cidx < nch; cidx += blockDim.x) {
            int idx = cidx << 3;
            *(uint4*)&sW[idx] = *(const uint4*)&wb[(size_t)cobase * KT + idx];
        }
#endif
    }
    __syncthreads();

    const int lane = threadIdx.x & 31;
    const int wid  = threadIdx.x >> 5;
    const int tbase0 = (blockIdx.x * CONV_WAVES + wid) * 64;
    if (tbase0 >= T) return;

    const int b  = blockIdx.z;
    const int n  = lane & 15;      // column (t) / A row (m)
    const int hi = lane >> 4;      // lane-group select
    const int t0 = tbase0 + n;

    v8f acc0 = {}, acc1 = {}, acc2 = {}, acc3 = {};
    const int step16 = 16 * S;

    for (int kk = 0; kk < KT; kk += 32) {
        // A operand: row m=n, elems [kk+16*hi .. +15]
        BfPack A;
        A.q[0] = *(const uint4*)&sW[(size_t)n * KT + kk + 16 * hi];
        A.q[1] = *(const uint4*)&sW[(size_t)n * KT + kk + 16 * hi + 8];
        // B operand: K = kk+16*hi .. +15 == 16 consecutive ci at fixed tap dk
        int kb  = kk + 16 * hi;
        int dk  = kb >> 9;
        int ci0 = kb & 511;
        int posbase = t0 * S - P + dk;

#define CONV_STEP(U, ACC)                                                           \
        {                                                                           \
            int pos = posbase + (U) * step16;                                       \
            BfPack Bv;                                                              \
            if (pos >= 0 && pos < Lin) {                                            \
                const uint4* pb =                                                   \
                    (const uint4*)&in[(((size_t)b * Lin) + pos) * CCH + ci0];       \
                Bv.q[0] = pb[0]; Bv.q[1] = pb[1];                                   \
            } else {                                                                \
                Bv.q[0] = make_uint4(0, 0, 0, 0);                                   \
                Bv.q[1] = make_uint4(0, 0, 0, 0);                                   \
            }                                                                       \
            ACC = __builtin_amdgcn_wmma_f32_16x16x32_bf16(                          \
                false, A.v, false, Bv.v, (short)0, ACC, false, false);              \
        }
        CONV_STEP(0, acc0)
        CONV_STEP(1, acc1)
        CONV_STEP(2, acc2)
        CONV_STEP(3, acc3)
#undef CONV_STEP
    }

    const int co0 = cobase + 8 * hi;
    float bs[8];
#pragma unroll
    for (int r = 0; r < 8; ++r) bs[r] = bias[co0 + r];

#define CONV_OUT(U, ACC)                                                            \
    {                                                                               \
        int t = t0 + (U) * 16;                                                      \
        if (t < T) {                                                                \
            unsigned short o[8];                                                    \
            _Pragma("unroll")                                                       \
            for (int r = 0; r < 8; ++r) o[r] = f2bf(ACC[r] + bs[r]);                \
            *(uint4*)&out[(((size_t)b * T) + t) * CCH + co0] = *(const uint4*)o;    \
        }                                                                           \
    }
    CONV_OUT(0, acc0)
    CONV_OUT(1, acc1)
    CONV_OUT(2, acc2)
    CONV_OUT(3, acc3)
#undef CONV_OUT
}

// -------------------- per-batch sum/sumsq over bf16 act --------------------
__global__ void k_stats(const unsigned short* __restrict__ act, float* __restrict__ st, int nPerB) {
    __shared__ float ss[256], sq[256];
    int b = blockIdx.y;
    const unsigned short* p = act + (size_t)b * nPerB;
    float s = 0.f, q = 0.f;
    for (long long i = (long long)blockIdx.x * blockDim.x + threadIdx.x; i < nPerB;
         i += (long long)gridDim.x * blockDim.x) {
        float v = bf2f(p[i]); s += v; q += v * v;
    }
    ss[threadIdx.x] = s; sq[threadIdx.x] = q;
    __syncthreads();
    for (int o = 128; o > 0; o >>= 1) {
        if ((int)threadIdx.x < o) { ss[threadIdx.x] += ss[threadIdx.x + o]; sq[threadIdx.x] += sq[threadIdx.x + o]; }
        __syncthreads();
    }
    if (threadIdx.x == 0) { atomicAdd(&st[2 + 2 * b], ss[0]); atomicAdd(&st[3 + 2 * b], sq[0]); }
}

// ---------------------- GroupNorm(1) + exact GELU, in place ----------------
__global__ void k_gn_gelu(unsigned short* __restrict__ act, const float* __restrict__ st,
                          const float* __restrict__ gamma, const float* __restrict__ beta, int T) {
    long long nPerB = (long long)T * CCH;
    long long N = (long long)BATCH * nPerB;
    long long i = (long long)blockIdx.x * blockDim.x + threadIdx.x;
    if (i >= N) return;
    int b = (int)(i / nPerB);
    int c = (int)(i & 511);
    float cnt  = (float)nPerB;
    float mean = st[2 + 2 * b] / cnt;
    float var  = st[3 + 2 * b] / cnt - mean * mean;
    float rstd = rsqrtf(var + 1e-5f);
    float h  = bf2f(act[i]);
    float hn = (h - mean) * rstd * gamma[c] + beta[c];
    float g  = 0.5f * hn * (1.0f + erff(hn * 0.70710678118654752f));
    act[i] = f2bf(g);
}

// ----------------------------- mask generation -----------------------------
__global__ void k_mask(unsigned char* __restrict__ mask, int* __restrict__ counts) {
    for (int i = threadIdx.x; i < BATCH * F_OUT; i += blockDim.x) mask[i] = 0;
    __syncthreads();
    if ((int)threadIdx.x < BATCH * NUM_M) {
        int b = threadIdx.x / NUM_M, j = threadIdx.x % NUM_M;
        int start = (int)(hash3(0x5eed5eedu, (unsigned)b, (unsigned)j) % (unsigned)(F_OUT - 10 + 1));
        for (int l = 0; l < 10; ++l) mask[b * F_OUT + start + l] = 1;
    }
    __syncthreads();
    if ((int)threadIdx.x < BATCH) {
        int c = 0;
        for (int f = 0; f < F_OUT; ++f) c += mask[threadIdx.x * F_OUT + f];
        counts[threadIdx.x] = c;
    }
}

// --------------------- top-k(100) of uniform scores per row ----------------
__global__ void k_topk(const unsigned char* __restrict__ mask, int* __restrict__ negidx) {
    __shared__ float sc[F_OUT];
    int r = blockIdx.x;                 // r = b*M_MAX + m
    int b = r / M_MAX, m = r % M_MAX;
    for (int f = threadIdx.x; f < F_OUT; f += 32)
        sc[f] = mask[b * F_OUT + f] ? -1e30f
                                    : u01(hash3(0xabcd1234u ^ (unsigned)b, (unsigned)m, (unsigned)f));
    __syncthreads();
    for (int j = 0; j < K_NEG; ++j) {
        float bv = -2e30f; int bi = 0;
        for (int f = threadIdx.x; f < F_OUT; f += 32) {
            float v = sc[f];
            if (v > bv) { bv = v; bi = f; }
        }
        for (int o = 16; o > 0; o >>= 1) {
            float ov = __shfl_xor(bv, o, 32);
            int   oi = __shfl_xor(bi, o, 32);
            if (ov > bv) { bv = ov; bi = oi; }
        }
        if (threadIdx.x == 0) { negidx[r * K_NEG + j] = bi; sc[bi] = -2e30f; }
        __syncthreads();
    }
}

// ------------- masked_features + features + mask -> d_out ------------------
__global__ void k_features_out(const unsigned short* __restrict__ act,
                               const unsigned char* __restrict__ mask,
                               const float* __restrict__ memb, float* __restrict__ out) {
    const int N = BATCH * CCH * F_OUT;
    int i = blockIdx.x * blockDim.x + threadIdx.x;
    if (i >= N) return;
    int t = i % F_OUT;
    int c = (i / F_OUT) % CCH;
    int b = i / (F_OUT * CCH);
    float f = bf2f(act[(((size_t)b * F_OUT) + t) * CCH + c]);
    out[OFF_FEAT + i] = f;
    bool mk = mask[b * F_OUT + t] != 0;
    out[OFF_MASKED + i] = mk ? memb[c] : f;
    if (c == 0) out[OFF_MASK + b * F_OUT + t] = mk ? 1.0f : 0.0f;
}

// --------------------------- negatives gather ------------------------------
__global__ void k_negatives(const unsigned short* __restrict__ act,
                            const int* __restrict__ negidx, const int* __restrict__ counts,
                            float* __restrict__ out) {
    const int N = BATCH * M_MAX * K_NEG * CCH;
    int i = blockIdx.x * blockDim.x + threadIdx.x;
    if (i >= N) return;
    int c  = i & 511;
    int j  = (i >> 9) % K_NEG;
    int rm = i / (CCH * K_NEG);          // b*M_MAX + m
    int m  = rm % M_MAX, b = rm / M_MAX;
    int t  = negidx[rm * K_NEG + j];
    float v = (m < counts[b]) ? bf2f(act[(((size_t)b * F_OUT) + t) * CCH + c]) : 0.0f;
    out[OFF_NEG + i] = v;
}

// ---------------------------------------------------------------------------
extern "C" void kernel_launch(void* const* d_in, const int* in_sizes, int n_in,
                              void* d_out, int out_size, void* d_ws, size_t ws_size,
                              hipStream_t stream) {
    (void)in_sizes; (void)n_in; (void)out_size; (void)ws_size;
    const float* x = (const float*)d_in[0];
    const float* cw[7]; const float* cb[7]; const float* gm[7]; const float* bt[7];
    for (int i = 0; i < 7; ++i) {
        cw[i] = (const float*)d_in[1 + i];
        cb[i] = (const float*)d_in[8 + i];
        gm[i] = (const float*)d_in[15 + i];
        bt[i] = (const float*)d_in[22 + i];
    }
    const float* memb = (const float*)d_in[29];
    float* out = (float*)d_out;
    char* ws = (char*)d_ws;

    // workspace layout (bytes)
    const size_t O_ACTA  = 0;                          // 262,152,192 (act1/3/5/7)
    const size_t O_ACTB  = 262152192ull;               // 131,080,192 (act2/4/6)
    const size_t O_W     = O_ACTB + 131080192ull;      //   1,572,864 (bf16 weights)
    const size_t O_ST    = O_W + 1572864ull;           //   stats: [0..1]=x, [2..17]=per-b
    const size_t O_MASK  = O_ST + 256;
    const size_t O_CNT   = O_MASK + 4096;
    const size_t O_NEGI  = O_CNT + 256;

    unsigned short* actA   = (unsigned short*)(ws + O_ACTA);
    unsigned short* actB   = (unsigned short*)(ws + O_ACTB);
    unsigned short* wbuf   = (unsigned short*)(ws + O_W);
    float*          st     = (float*)(ws + O_ST);
    unsigned char*  maskb  = (unsigned char*)(ws + O_MASK);
    int*            counts = (int*)(ws + O_CNT);
    int*            negidx = (int*)(ws + O_NEGI);

    const int Ls[8] = {160000, 32001, 16001, 8001, 4001, 2001, 1001, 501};
    const int Ks[7] = {10, 3, 3, 3, 3, 2, 2};
    const int Ss[7] = {5, 2, 2, 2, 2, 2, 2};
    const int Ps[7] = {5, 1, 1, 1, 1, 1, 1};

    // ---- input normalization stats ----
    k_zero_f32<<<1, 32, 0, stream>>>(st, 18);
    k_reduce_x<<<1024, 256, 0, stream>>>(x, st, 1280000LL);

    // ---- layer 0 (cin=1, direct) ----
    {
        long long N = (long long)BATCH * Ls[1] * CCH;
        k_conv1<<<(unsigned)((N + 255) / 256), 256, 0, stream>>>(x, cw[0], cb[0], st, actA);
        k_stats<<<dim3(512, BATCH), 256, 0, stream>>>(actA, st, Ls[1] * CCH);
        k_gn_gelu<<<(unsigned)((N + 255) / 256), 256, 0, stream>>>(actA, st, gm[0], bt[0], Ls[1]);
    }

    // ---- layers 1..6 (cin=512, WMMA implicit GEMM) ----
    unsigned short* cur = actA;
    unsigned short* nxt = actB;
    for (int i = 1; i < 7; ++i) {
        int Lin = Ls[i], T = Ls[i + 1];
        int KT = CCH * Ks[i];
        k_wprep<<<(CCH * KT + 255) / 256, 256, 0, stream>>>(cw[i], wbuf, Ks[i], KT);
        int tgroups = (T + 63) / 64;                       // 64 t per wave
        dim3 g((tgroups + CONV_WAVES - 1) / CONV_WAVES, CCH / 16, BATCH);
        k_conv_wmma<<<g, 256, 0, stream>>>(cur, wbuf, cb[i], nxt, Lin, T, Ss[i], Ps[i], KT);
        k_zero_f32<<<1, 32, 0, stream>>>(st + 2, 16);
        k_stats<<<dim3(512, BATCH), 256, 0, stream>>>(nxt, st, T * CCH);
        long long N = (long long)BATCH * T * CCH;
        k_gn_gelu<<<(unsigned)((N + 255) / 256), 256, 0, stream>>>(nxt, st, gm[i], bt[i], T);
        unsigned short* tmp = cur; cur = nxt; nxt = tmp;
    }
    // cur now holds final features, bf16 [8][501][512]

    // ---- masking, top-k negatives, outputs ----
    k_mask<<<1, 256, 0, stream>>>(maskb, counts);
    k_topk<<<BATCH * M_MAX, 32, 0, stream>>>(maskb, negidx);
    {
        int N = BATCH * CCH * F_OUT;
        k_features_out<<<(N + 255) / 256, 256, 0, stream>>>(cur, maskb, memb, out);
    }
    {
        int N = BATCH * M_MAX * K_NEG * CCH;
        k_negatives<<<(N + 255) / 256, 256, 0, stream>>>(cur, negidx, counts, out);
    }
}
